// Decoder_4217657885137
// MI455X (gfx1250) — matrix operand order repaired and checked by reading, not compiled, not run on hardware
//
#include <hip/hip_runtime.h>
#include <cstdint>
#include <cstddef>

// ---------------------------------------------------------------------------
// Types
// ---------------------------------------------------------------------------
typedef unsigned short bf_t;  // bf16 storage (host-safe)
typedef __attribute__((ext_vector_type(16))) __bf16 v16bf;
typedef __attribute__((ext_vector_type(8)))  __bf16 v8bf;
typedef __attribute__((ext_vector_type(8)))  float  v8f;

#define HID   1024
#define BATCH 64
#define IN0   1536
#define OUTP  512
#define SEQL  512
#define KB    32       // K-slab per WMMA (16x16x32 bf16)

// ---------------------------------------------------------------------------
// Device helpers
// ---------------------------------------------------------------------------
__device__ __forceinline__ unsigned f2bf_bits(float f) {
    unsigned u = __float_as_uint(f);
    return (u + 0x7FFFu + ((u >> 16) & 1u)) >> 16;   // round-to-nearest-even
}
__device__ __forceinline__ float sigm(float x) { return 1.0f / (1.0f + __expf(-x)); }

// Async global->LDS copies (CDNA5 GLOBAL_LOAD_ASYNC_TO_LDS_*, ASYNCcnt).
// Low 32 bits of a generic pointer into LDS are the wave-relative LDS offset.
__device__ __forceinline__ void async_ld_b128(void* lds, const void* g) {
    asm volatile("global_load_async_to_lds_b128 %0, %1, off"
                 :: "v"((unsigned)(uintptr_t)lds),
                    "v"((unsigned long long)(uintptr_t)g)
                 : "memory");
}
__device__ __forceinline__ void async_ld_b64(void* lds, const void* g) {
    asm volatile("global_load_async_to_lds_b64 %0, %1, off"
                 :: "v"((unsigned)(uintptr_t)lds),
                    "v"((unsigned long long)(uintptr_t)g)
                 : "memory");
}
__device__ __forceinline__ void wait_async0() {
    asm volatile("s_wait_asynccnt 0" ::: "memory");
}

// A fragment (16x32 bf16, M x K): lane L<16 holds row L, K = 0..15;
// lane L>=16 holds row L-16, K = 16..31.
__device__ __forceinline__ v16bf load_a_frag(const bf_t* row_base, int lane) {
    const bf_t* p = row_base + ((lane >= 16) ? 16 : 0);
    v8bf lo = *(const v8bf*)(p);        // K = base+0..7
    v8bf hi = *(const v8bf*)(p + 8);    // K = base+8..15
    return __builtin_shufflevector(lo, hi, 0,1,2,3,4,5,6,7,8,9,10,11,12,13,14,15);
}
// B fragment (32x16 bf16, K x N) from W tile stored [64 rows(n)][32 cols(k)]:
// lane L<16 holds column L with K=0..15; lane L>=16 holds column L-16, K=16..31.
__device__ __forceinline__ v16bf load_b_frag(const bf_t* wtile, int n_tile, int lane) {
    const bf_t* p = wtile + (size_t)(n_tile * 16 + (lane & 15)) * 32 + ((lane < 16) ? 0 : 16);
    return *(const v16bf*)(p);
}
__device__ __forceinline__ v8f wmma_bf16(v16bf a, v16bf b, v8f c) {
    return __builtin_amdgcn_wmma_f32_16x16x32_bf16(false, a, false, b, (short)0, c, false, false);
}

// ---------------------------------------------------------------------------
// Fused LSTM layer step: gates = X@Wih^T + Hprev@Whh^T (+biases), plus
// h2pre = Hskip@Wh2 (Wh2 pre-transposed), then full cell + mask epilogue.
// grid = HID/64 blocks, block = 512 threads (16 wave32s).
// Double-buffered LDS; weights + bf16 activations staged via async-to-LDS.
// ---------------------------------------------------------------------------
template <bool XBF16, int KIN>
__global__ __launch_bounds__(512, 1)
void lstm_layer_kernel(const void* __restrict__ Xin,
                       const bf_t* __restrict__ Wih,  const bf_t* __restrict__ Whh,
                       const bf_t* __restrict__ Wh2T,
                       const bf_t* __restrict__ Hprev, const bf_t* __restrict__ Hskip,
                       const float* __restrict__ b_ih, const float* __restrict__ b_hh,
                       const float* __restrict__ bh2,  const int* __restrict__ mask,
                       float* __restrict__ Cst, bf_t* __restrict__ Hout)
{
    __shared__ __align__(16) bf_t sA[2][BATCH][KB + 8];   // padded rows (80B stride)
    __shared__ __align__(16) bf_t sW[2][4][64][KB];

    const int tid = threadIdx.x, lane = tid & 31, wv = tid >> 5;
    const int m_tile = wv & 3, n_tile = wv >> 2;          // 4x4 16x16 tiles over 64x64
    const int n0 = blockIdx.x * 64;
    const int arow = m_tile * 16 + (lane & 15);
    const int am = tid >> 3, ac = tid & 7;                // A staging: 8 chunks/row

    v8f acc[5];
    #pragma unroll
    for (int g = 0; g < 5; ++g) acc[g] = (v8f){0.f,0.f,0.f,0.f,0.f,0.f,0.f,0.f};

    const float* Xf = (const float*)Xin;
    const bf_t*  Xb = (const bf_t*)Xin;

    // Staging helpers (capture-free, inlined) -------------------------------
    auto stage_w4 = [&](int buf, const bf_t* W, int ld, int k0) {
        #pragma unroll
        for (int j = 0; j < 2; ++j) {                     // 1024 x b128 / 512 thr
            const int i = tid + j * 512;
            const int g = i >> 7, r = (i >> 1) & 63, c = i & 1;
            async_ld_b128(&sW[buf][g][r][c * 8],
                          W + (size_t)(g * HID + n0 + r) * ld + k0 + c * 8);
        }
    };
    auto stage_w1 = [&](int buf, const bf_t* W, int k0) { // one 64x32 tile, b64/thr
        const int r = tid >> 3, c = tid & 7;
        async_ld_b64(&sW[buf][0][r][c * 4],
                     W + (size_t)(n0 + r) * HID + k0 + c * 4);
    };
    auto stage_a_async = [&](int buf, const bf_t* src, int ld, int k0) {
        async_ld_b64(&sA[buf][am][ac * 4], src + (size_t)am * ld + k0 + ac * 4);
    };
    auto store_a_cvt = [&](int buf, float4 f) {
        uint2 p;
        p.x = f2bf_bits(f.x) | (f2bf_bits(f.y) << 16);
        p.y = f2bf_bits(f.z) | (f2bf_bits(f.w) << 16);
        *(uint2*)&sA[buf][am][ac * 4] = p;
    };
    auto compute4 = [&](int buf) {
        v16bf a = load_a_frag(&sA[buf][arow][0], lane);
        #pragma unroll
        for (int g = 0; g < 4; ++g)
            acc[g] = wmma_bf16(a, load_b_frag(&sW[buf][g][0][0], n_tile, lane), acc[g]);
    };

    // ---- Phase 1: gates += X @ Wih^T  (double-buffered, pipelined) --------
    {
        int p = 0;
        stage_w4(p, Wih, KIN, 0);
        if (XBF16) stage_a_async(p, Xb, KIN, 0);
        else store_a_cvt(p, *(const float4*)(Xf + (size_t)am * KIN + ac * 4));
        wait_async0();
        __syncthreads();
        for (int k0 = 0; k0 < KIN; k0 += KB) {
            const int nb = p ^ 1, kn = k0 + KB;
            const bool more = kn < KIN;
            float4 fnext;
            if (more) {
                stage_w4(nb, Wih, KIN, kn);
                if (XBF16) stage_a_async(nb, Xb, KIN, kn);
                else fnext = *(const float4*)(Xf + (size_t)am * KIN + kn + ac * 4);
            }
            compute4(p);
            if (more && !XBF16) store_a_cvt(nb, fnext);
            wait_async0();
            __syncthreads();
            p = nb;
        }
    }

    // ---- Phase 2: gates += Hprev @ Whh^T ----------------------------------
    {
        int p = 0;
        stage_w4(p, Whh, HID, 0);
        stage_a_async(p, Hprev, HID, 0);
        wait_async0();
        __syncthreads();
        for (int k0 = 0; k0 < HID; k0 += KB) {
            const int nb = p ^ 1, kn = k0 + KB;
            if (kn < HID) {
                stage_w4(nb, Whh, HID, kn);
                stage_a_async(nb, Hprev, HID, kn);
            }
            compute4(p);
            wait_async0();
            __syncthreads();
            p = nb;
        }
    }

    // ---- Phase 3: h2pre = Hskip @ Wh2 (Wh2T stored as [n][k]) -------------
    {
        int p = 0;
        stage_w1(p, Wh2T, 0);
        stage_a_async(p, Hskip, HID, 0);
        wait_async0();
        __syncthreads();
        for (int k0 = 0; k0 < HID; k0 += KB) {
            const int nb = p ^ 1, kn = k0 + KB;
            if (kn < HID) {
                stage_w1(nb, Wh2T, kn);
                stage_a_async(nb, Hskip, HID, kn);
            }
            v16bf a = load_a_frag(&sA[p][arow][0], lane);
            acc[4] = wmma_bf16(a, load_b_frag(&sW[p][0][0][0], n_tile, lane), acc[4]);
            wait_async0();
            __syncthreads();
            p = nb;
        }
    }

    // ---- Epilogue: full LSTM cell + skip mix, register-resident -----------
    // C/D layout: VGPR r -> M = m_tile*16 + r + (lane>=16 ? 8 : 0), N = lane&15.
    const int col = n0 + n_tile * 16 + (lane & 15);
    const int mbase = m_tile * 16 + ((lane >= 16) ? 8 : 0);
    const float bi0 = b_ih[col]           + b_hh[col];
    const float bi1 = b_ih[HID + col]     + b_hh[HID + col];
    const float bi2 = b_ih[2 * HID + col] + b_hh[2 * HID + col];
    const float bi3 = b_ih[3 * HID + col] + b_hh[3 * HID + col];
    const float b2  = bh2[col];
    const int   code = mask[col];
    const float m1 = (code == 0 || code == 2) ? 1.f : 0.f;
    const float m2 = (code == 1 || code == 2) ? 1.f : 0.f;
    #pragma unroll
    for (int r = 0; r < 8; ++r) {
        const int m = mbase + r;
        float ig = sigm(acc[0][r] + bi0);
        float fg = sigm(acc[1][r] + bi1);
        float gg = tanhf(acc[2][r] + bi2);
        float og = sigm(acc[3][r] + bi3);
        float cold = Cst[(size_t)m * HID + col];
        float cnew = fg * cold + ig * gg;
        float h1 = og * tanhf(cnew);
        float h2 = sigm(acc[4][r] + b2);
        Cst[(size_t)m * HID + col]  = cnew;
        Hout[(size_t)m * HID + col] = (bf_t)(f2bf_bits(h1 * m1 + h2 * m2));
    }
}

// ---------------------------------------------------------------------------
// Output projection: Y = Hin(64x1024,bf16) @ Wout^T + b_out. grid = OUTP/64.
// ---------------------------------------------------------------------------
__global__ __launch_bounds__(512, 1)
void out_proj_kernel(const bf_t* __restrict__ Hin, const bf_t* __restrict__ Wout,
                     const float* __restrict__ bout, float* __restrict__ Y)
{
    __shared__ __align__(16) bf_t sA[2][BATCH][KB + 8];
    __shared__ __align__(16) bf_t sW[2][64][KB];
    const int tid = threadIdx.x, lane = tid & 31, wv = tid >> 5;
    const int m_tile = wv & 3, n_tile = wv >> 2;
    const int n0 = blockIdx.x * 64;
    const int arow = m_tile * 16 + (lane & 15);
    const int am = tid >> 3, ac = tid & 7;

    auto stage = [&](int buf, int k0) {
        async_ld_b64(&sA[buf][am][ac * 4], Hin + (size_t)am * HID + k0 + ac * 4);
        const int r = tid >> 3, c = tid & 7;
        async_ld_b64(&sW[buf][r][c * 4], Wout + (size_t)(n0 + r) * HID + k0 + c * 4);
    };

    v8f acc = (v8f){0.f,0.f,0.f,0.f,0.f,0.f,0.f,0.f};
    int p = 0;
    stage(p, 0);
    wait_async0();
    __syncthreads();
    for (int k0 = 0; k0 < HID; k0 += KB) {
        const int nb = p ^ 1, kn = k0 + KB;
        if (kn < HID) stage(nb, kn);
        v16bf a = load_a_frag(&sA[p][arow][0], lane);
        acc = wmma_bf16(a, load_b_frag(&sW[p][0][0], n_tile, lane), acc);
        wait_async0();
        __syncthreads();
        p = nb;
    }
    const int col = n0 + n_tile * 16 + (lane & 15);
    const int mbase = m_tile * 16 + ((lane >= 16) ? 8 : 0);
    const float bb = bout[col];
    #pragma unroll
    for (int r = 0; r < 8; ++r)
        Y[(size_t)(mbase + r) * OUTP + col] = acc[r] + bb;
}

// ---------------------------------------------------------------------------
// Precompute kernels
// ---------------------------------------------------------------------------
__global__ void cvt_bf16_kernel(const float* __restrict__ in, bf_t* __restrict__ out, int n4) {
    int i = blockIdx.x * 256 + threadIdx.x;            // n4 = n/4 float4 chunks
    if (i < n4) {
        float4 f = *(const float4*)(in + (size_t)i * 4);
        uint2 p;
        p.x = f2bf_bits(f.x) | (f2bf_bits(f.y) << 16);
        p.y = f2bf_bits(f.z) | (f2bf_bits(f.w) << 16);
        *(uint2*)(out + (size_t)i * 4) = p;
    }
}
// Wh2 (H x H) used as h @ Wh2 -> store transposed so B-tiles are row-major [n][k].
__global__ void cvt_transpose_kernel(const float* __restrict__ in, bf_t* __restrict__ out) {
    int i = blockIdx.x * 256 + threadIdx.x;
    if (i < HID * HID) {
        int nI = i >> 10, k = i & (HID - 1);
        out[i] = (bf_t)f2bf_bits(in[(size_t)k * HID + nI]);
    }
}
__global__ void init_layer_kernel(const float* __restrict__ h0l, const float* __restrict__ c0l,
                                  float* __restrict__ cst, bf_t* __restrict__ slot1,
                                  bf_t* __restrict__ slot2, int n) {
    int i = blockIdx.x * 256 + threadIdx.x;
    if (i < n) {
        cst[i]   = c0l[i];
        slot2[i] = (bf_t)f2bf_bits(h0l[i]);  // h(t=-1) = h0
        slot1[i] = (bf_t)0;                  // h(t=-2) = 0 (SKIP pre-warm-up)
    }
}

// ---------------------------------------------------------------------------
// Host
// ---------------------------------------------------------------------------
extern "C" void kernel_launch(void* const* d_in, const int* in_sizes, int n_in,
                              void* d_out, int out_size, void* d_ws, size_t ws_size,
                              hipStream_t stream)
{
    const float* targets = (const float*)d_in[0];
    const float* h0      = (const float*)d_in[1];
    const float* c0      = (const float*)d_in[2];
    const int*   mask    = (const int*)d_in[3];
    const float* W_ih0   = (const float*)d_in[4];
    const float* W_hh0   = (const float*)d_in[5];
    const float* b_ih0   = (const float*)d_in[6];
    const float* b_hh0   = (const float*)d_in[7];
    const float* W_ih1   = (const float*)d_in[8];
    const float* W_hh1   = (const float*)d_in[9];
    const float* b_ih1   = (const float*)d_in[10];
    const float* b_hh1   = (const float*)d_in[11];
    const float* Wh2_0   = (const float*)d_in[12];
    const float* bh2_0   = (const float*)d_in[13];
    const float* Wh2_1   = (const float*)d_in[14];
    const float* bh2_1   = (const float*)d_in[15];
    const float* W_outp  = (const float*)d_in[16];
    const float* b_outp  = (const float*)d_in[17];
    float* out = (float*)d_out;

    char* base = (char*)d_ws;
    size_t off = 0;
    auto abf = [&](size_t elems) -> bf_t* {
        bf_t* p = (bf_t*)(base + off);
        off = (off + elems * sizeof(bf_t) + 255) & ~(size_t)255;
        return p;
    };
    bf_t* wih0  = abf((size_t)4 * HID * IN0);
    bf_t* whh0  = abf((size_t)4 * HID * HID);
    bf_t* wih1  = abf((size_t)4 * HID * HID);
    bf_t* whh1  = abf((size_t)4 * HID * HID);
    bf_t* wh2t0 = abf((size_t)HID * HID);
    bf_t* wh2t1 = abf((size_t)HID * HID);
    bf_t* woutb = abf((size_t)OUTP * HID);
    bf_t* hbuf[2][3];
    for (int l = 0; l < 2; ++l)
        for (int s = 0; s < 3; ++s) hbuf[l][s] = abf((size_t)BATCH * HID);
    float* cst = (float*)(base + off);
    off = (off + (size_t)2 * BATCH * HID * sizeof(float) + 255) & ~(size_t)255;

    // ---- One-time precompute (every call; deterministic) ------------------
    auto cvt = [&](const float* in, bf_t* o, int n) {
        int n4 = n / 4;
        cvt_bf16_kernel<<<(n4 + 255) / 256, 256, 0, stream>>>(in, o, n4);
    };
    cvt(W_ih0, wih0, 4 * HID * IN0);
    cvt(W_hh0, whh0, 4 * HID * HID);
    cvt(W_ih1, wih1, 4 * HID * HID);
    cvt(W_hh1, whh1, 4 * HID * HID);
    cvt(W_outp, woutb, OUTP * HID);
    cvt_transpose_kernel<<<(HID * HID + 255) / 256, 256, 0, stream>>>(Wh2_0, wh2t0);
    cvt_transpose_kernel<<<(HID * HID + 255) / 256, 256, 0, stream>>>(Wh2_1, wh2t1);
    const int NS = BATCH * HID;
    init_layer_kernel<<<(NS + 255) / 256, 256, 0, stream>>>(h0, c0, cst,
                                                            hbuf[0][1], hbuf[0][2], NS);
    init_layer_kernel<<<(NS + 255) / 256, 256, 0, stream>>>(h0 + NS, c0 + NS, cst + NS,
                                                            hbuf[1][1], hbuf[1][2], NS);

    // ---- Recurrent loop: 3 kernels per step -------------------------------
    for (int t = 0; t < SEQL; ++t) {
        const int sw = t % 3;            // slot written: h(t)
        const int sp = (t + 2) % 3;      // h(t-1) for W_hh
        const int ss = (t + 1) % 3;      // h(t-2) for Wh2 (SKIP=1 skip buffer)

        lstm_layer_kernel<false, IN0><<<HID / 64, 512, 0, stream>>>(
            (const void*)(targets + (size_t)t * BATCH * IN0),
            wih0, whh0, wh2t0, hbuf[0][sp], hbuf[0][ss],
            b_ih0, b_hh0, bh2_0, mask + (size_t)(t * 2 + 0) * HID,
            cst, hbuf[0][sw]);

        lstm_layer_kernel<true, HID><<<HID / 64, 512, 0, stream>>>(
            (const void*)hbuf[0][sw],
            wih1, whh1, wh2t1, hbuf[1][sp], hbuf[1][ss],
            b_ih1, b_hh1, bh2_1, mask + (size_t)(t * 2 + 1) * HID,
            cst + NS, hbuf[1][sw]);

        out_proj_kernel<<<OUTP / 64, 512, 0, stream>>>(
            hbuf[1][sw], woutb, b_outp, out + (size_t)t * BATCH * OUTP);
    }
}
